// Mog_12790412607597
// MI455X (gfx1250) — matrix operand order repaired and checked by reading, not compiled, not run on hardware
//
#include <hip/hip_runtime.h>
#include <hip/hip_bf16.h>
#include <math.h>

// ---------------------------------------------------------------------------
// Mogrifier bi-GRU for MI455X (gfx1250).
//   Phase 0: embedding gather -> f16 xT [T,B,D]
//   Phase 1: weights -> f16, pre-swizzled into WMMA B-fragment layout
//   Phase 2: persistent scan kernel, 1 workgroup per direction, all GEMMs on
//            v_wmma_f32_16x16x32_f16, state resident in LDS, weights stream
//            from L2 (k-loops kept rolled so LICM cannot hoist+spill them),
//            gate nonlinearities on the native v_rcp/v_exp path
//   Phase 3: gather + gated attention head (small, plain VALU)
// Workspace: ~170 MB (xT 33.5MB, weights 2MB, out_f/out_b 67MB each)
// ---------------------------------------------------------------------------

typedef __attribute__((ext_vector_type(16))) _Float16 v16h;
typedef __attribute__((ext_vector_type(8)))  _Float16 v8h;
typedef __attribute__((ext_vector_type(8)))  float    v8f;

#define DEVINL __device__ __forceinline__

// Fast nonlinearities on the serial critical path: v_exp_f32 + v_rcp_f32,
// no IEEE divide expansion.
DEVINL float fast_sig(float x) {
  return __builtin_amdgcn_rcpf(1.0f + __expf(-x));
}
DEVINL float fast_tanh(float x) {
  // tanh(x) = 1 - 2/(1+exp(2x))
  return 1.0f - 2.0f * __builtin_amdgcn_rcpf(1.0f + __expf(2.0f * x));
}

DEVINL v8f wmma_f16(v16h a, v16h b, v8f c) {
  // 8 args: (neg_a, A, neg_b, B, c_mod, C, reuse_a, reuse_b)
  return __builtin_amdgcn_wmma_f32_16x16x32_f16(false, a, false, b, (short)0, c,
                                                false, false);
}

DEVINL v8f zero8() {
  v8f z;
#pragma unroll
  for (int i = 0; i < 8; ++i) z[i] = 0.0f;
  return z;
}

// A fragment (16x32 f16, contiguous k order, same as B): row M = lane&15.
// halves 0..15 -> K = kbase + (lane>=16 ? 16 : 0) + i
// Source matrix is row-major [64][256] f16 in LDS.
DEVINL v16h load_a_frag(const _Float16* mat, int mtile, int kbase, int lane) {
  int row = mtile * 16 + (lane & 15);
  int k0  = kbase + ((lane & 16) ? 16 : 0);
  const v8h* p = (const v8h*)(mat + row * 256 + k0);
  v8h lo = p[0];
  v8h hi = p[1];  // +8 halves
  v16h a;
#pragma unroll
  for (int i = 0; i < 8; ++i) { a[i] = lo[i]; a[8 + i] = hi[i]; }
  return a;
}

// B fragment (32x16 f16): N = lane&15, K = (lane>=16 ? 16:0) + i.
// Weights are pre-swizzled so each lane reads 16 contiguous halves (32B).
DEVINL v16h load_b_frag(const _Float16* frag, int lane) {
  const v8h* p = (const v8h*)(frag + lane * 16);
  v8h lo = p[0];
  v8h hi = p[1];
  v16h b;
#pragma unroll
  for (int i = 0; i < 8; ++i) { b[i] = lo[i]; b[8 + i] = hi[i]; }
  return b;
}

// One mogrifier round: dst[m][n] = 2*sigmoid((A @ Bsw)[m][n]) * dst[m][n]
// A = aL (LDS f16 [64][256]), Bsw = swizzled weights [256 -> 256].
// A wave owns 4 consecutive ntiles at a single mtile (= wave>>2), so one
// A-fragment load feeds 4 independent WMMA accumulator chains.
DEVINL void mog_round(const _Float16* aL, const _Float16* bsw, _Float16* dst,
                      int wave, int lane) {
  int mtile = wave >> 2;
  int nbase = (wave & 3) * 4;  // ntile = nbase + tix, tix in [0,4)
  const _Float16* b0 = bsw + ((size_t)nbase * 8 << 9);  // tile stride: 8 frags
  v8f acc0 = zero8(), acc1 = zero8(), acc2 = zero8(), acc3 = zero8();
#pragma unroll 1
  for (int ks = 0; ks < 8; ++ks) {
    v16h a = load_a_frag(aL, mtile, ks * 32, lane);
    const _Float16* bp = b0 + ((size_t)ks << 9);
    acc0 = wmma_f16(a, load_b_frag(bp,                      lane), acc0);
    acc1 = wmma_f16(a, load_b_frag(bp + ((size_t)8  << 9),  lane), acc1);
    acc2 = wmma_f16(a, load_b_frag(bp + ((size_t)16 << 9),  lane), acc2);
    acc3 = wmma_f16(a, load_b_frag(bp + ((size_t)24 << 9),  lane), acc3);
  }
  int m0 = mtile * 16 + ((lane & 16) ? 8 : 0);
  int nb = nbase * 16 + (lane & 15);
#pragma unroll
  for (int r = 0; r < 8; ++r) {
    int row = (m0 + r) * 256;
    float g0 = 2.0f * fast_sig(acc0[r]);
    float g1 = 2.0f * fast_sig(acc1[r]);
    float g2 = 2.0f * fast_sig(acc2[r]);
    float g3 = 2.0f * fast_sig(acc3[r]);
    dst[row + nb]      = (_Float16)(g0 * (float)dst[row + nb]);
    dst[row + nb + 16] = (_Float16)(g1 * (float)dst[row + nb + 16]);
    dst[row + nb + 32] = (_Float16)(g2 * (float)dst[row + nb + 32]);
    dst[row + nb + 48] = (_Float16)(g3 * (float)dst[row + nb + 48]);
  }
}

// ------------------------------ Phase 0 ------------------------------------
__global__ void __launch_bounds__(256)
embed_kernel(const int* __restrict__ sents, const float* __restrict__ emb,
             _Float16* __restrict__ xT) {
  const int B = 64, T = 1024, D = 256;
  int idx = blockIdx.x * blockDim.x + threadIdx.x;
  if (idx >= T * B * D) return;
  int d = idx & (D - 1);
  int b = (idx >> 8) & (B - 1);
  int t = idx >> 14;
  int tok = sents[b * T + t];
  xT[idx] = (_Float16)emb[(size_t)tok * D + d];
}

// ------------------------------ Phase 1 ------------------------------------
// Swizzle a [K][N] B-operand (optionally transposed source [N][K]) into
// fragment-major f16: frag = ntile*(K/32)+ks, then lane*16+i with
// k = ks*32 + (lane>=16?16:0) + i, n = ntile*16 + (lane&15).
__global__ void __launch_bounds__(256)
swizzle_b_kernel(const float* __restrict__ src, _Float16* __restrict__ dst,
                 int K, int N, int transpose) {
  int idx = blockIdx.x * blockDim.x + threadIdx.x;
  if (idx >= K * N) return;
  int i    = idx & 15;
  int lane = (idx >> 4) & 31;
  int frag = idx >> 9;
  int nK   = K >> 5;
  int ntile = frag / nK;
  int ks    = frag - ntile * nK;
  int k = ks * 32 + ((lane & 16) ? 16 : 0) + i;
  int n = ntile * 16 + (lane & 15);
  float v = transpose ? src[(size_t)n * K + k] : src[(size_t)k * N + n];
  dst[idx] = (_Float16)v;
}

// ------------------------------ Phase 2 ------------------------------------
__global__ void __launch_bounds__(512)
scan_kernel(const _Float16* __restrict__ xT,
            const _Float16* __restrict__ Qsw, const _Float16* __restrict__ Rsw,
            const _Float16* __restrict__ WihSw, const _Float16* __restrict__ WhhSw,
            const int* __restrict__ lens,
            const float* __restrict__ bih_f, const float* __restrict__ bhh_f,
            const float* __restrict__ bih_b, const float* __restrict__ bhh_b,
            float* __restrict__ out_f, float* __restrict__ out_b) {
  const int B = 64, T = 1024, H = 256;
  const int BD = 16384;  // B * 256
  int dir  = blockIdx.x;
  int tid  = threadIdx.x;
  int lane = tid & 31;
  int wave = tid >> 5;

  extern __shared__ char smem[];
  _Float16* x16  = (_Float16*)smem;  // mogrified x, f16 [64][256]
  _Float16* h16  = x16 + BD;         // mogrified h, f16 [64][256]
  _Float16* hc16 = h16 + BD;         // carry h (pre-mogrify), f16 [64][256]
  float*    bihL  = (float*)(hc16 + BD);  // [768]
  float*    bhhL  = bihL + 768;           // [768]
  int*      maskL = (int*)(bhhL + 768);   // [64]

  const float* bih = dir ? bih_b : bih_f;
  const float* bhh = dir ? bhh_b : bhh_f;
  const _Float16* Qw = Qsw   + (size_t)dir * 65536;
  const _Float16* Rw = Rsw   + (size_t)dir * 65536;
  const _Float16* Wi = WihSw + (size_t)dir * 196608;
  const _Float16* Wh = WhhSw + (size_t)dir * 196608;
  float* outG = dir ? out_b : out_f;

  for (int i = tid; i < 768; i += 512) { bihL[i] = bih[i]; bhhL[i] = bhh[i]; }
  for (int i = tid; i < BD; i += 512) hc16[i] = (_Float16)0.0f;
  __syncthreads();

#pragma unroll 1
  for (int t = 0; t < T; ++t) {
    int tt = dir ? (T - 1 - t) : t;

    // stage x_t into LDS, mirror carry h into working h16
    const _Float16* xsrc = xT + (size_t)tt * BD;
    for (int i = tid * 8; i < BD; i += 4096) {
      *(v8h*)(x16 + i) = *(const v8h*)(xsrc + i);
      *(v8h*)(h16 + i) = *(const v8h*)(hc16 + i);
    }
    if (tid < B) maskL[tid] = (tt < lens[tid]) ? 1 : 0;
    if (t + 1 < T) {  // prefetch next timestep's x (global_prefetch_b8)
      int tn = dir ? (T - 2 - t) : (t + 1);
      __builtin_prefetch(xT + (size_t)tn * BD + tid * 32, 0, 0);
    }
    __syncthreads();

    // mogrifier: x=2s(hQ)x; h=2s(xR)h; x=2s(hQ)x; h=2s(xR)h (all in place)
    mog_round(h16, Qw, x16, wave, lane); __syncthreads();
    mog_round(x16, Rw, h16, wave, lane); __syncthreads();
    mog_round(h16, Qw, x16, wave, lane); __syncthreads();
    mog_round(x16, Rw, h16, wave, lane); __syncthreads();

    // GRU gates: one (mtile, htile) output tile at a time, 6 accumulators
    // (i_r/i_z/i_n from x@Wih.T, h_r/h_z/h_n from h@Whh.T). A-fragments are
    // shared by all 6 chains; B streams from L2. Loops kept rolled so the
    // compiler cannot hoist the weight loads out of the time loop and spill.
    int mtile = wave >> 2;
#pragma unroll 1
    for (int tix = 0; tix < 4; ++tix) {
      int htile = (wave & 3) * 4 + tix;
      const _Float16* bi = Wi + ((size_t)htile * 8 << 9);
      const _Float16* bh = Wh + ((size_t)htile * 8 << 9);
      v8f air = zero8(), aiz = zero8(), ain = zero8();
      v8f ahr = zero8(), ahz = zero8(), ahn = zero8();
#pragma unroll 1
      for (int ks = 0; ks < 8; ++ks) {
        v16h ax = load_a_frag(x16, mtile, ks * 32, lane);
        v16h ah = load_a_frag(h16, mtile, ks * 32, lane);
        const _Float16* bik = bi + ((size_t)ks << 9);
        const _Float16* bhk = bh + ((size_t)ks << 9);
        // gate stride: 16 ntiles * 8 ks-frags * 512 halves = 65536 halves
        air = wmma_f16(ax, load_b_frag(bik,                       lane), air);
        aiz = wmma_f16(ax, load_b_frag(bik + (size_t)65536,       lane), aiz);
        ain = wmma_f16(ax, load_b_frag(bik + (size_t)131072,      lane), ain);
        ahr = wmma_f16(ah, load_b_frag(bhk,                       lane), ahr);
        ahz = wmma_f16(ah, load_b_frag(bhk + (size_t)65536,       lane), ahz);
        ahn = wmma_f16(ah, load_b_frag(bhk + (size_t)131072,      lane), ahn);
      }
      int m0 = mtile * 16 + ((lane & 16) ? 8 : 0);
      int n  = htile * 16 + (lane & 15);
      float b_ir = bihL[n],       b_hr = bhhL[n];
      float b_iz = bihL[256 + n], b_hz = bhhL[256 + n];
      float b_in = bihL[512 + n], b_hn = bhhL[512 + n];
#pragma unroll
      for (int r = 0; r < 8; ++r) {
        int m = m0 + r;
        float rr = fast_sig(air[r] + b_ir + ahr[r] + b_hr);
        float zz = fast_sig(aiz[r] + b_iz + ahz[r] + b_hz);
        float nn = fast_tanh(ain[r] + b_in + rr * (ahn[r] + b_hn));
        float hm = (float)h16[m * H + n];          // mogrified h (z-blend)
        float hnew = (1.0f - zz) * nn + zz * hm;
        float hold = (float)hc16[m * H + n];       // carry h (mask-blend)
        int   mk = maskL[m];
        hc16[m * H + n] = (_Float16)(mk ? hnew : hold);
        outG[((size_t)tt * B + m) * H + n] = mk ? hnew : 0.0f;
      }
    }
    __syncthreads();
  }
}

// ------------------------------ Phase 3 ------------------------------------
DEVINL float blk_max(float v, float* red) {
  int tid = threadIdx.x;
  red[tid] = v; __syncthreads();
  for (int s = 128; s > 0; s >>= 1) {
    if (tid < s) red[tid] = fmaxf(red[tid], red[tid + s]);
    __syncthreads();
  }
  float r = red[0]; __syncthreads();
  return r;
}

DEVINL float blk_sum(float v, float* red) {
  int tid = threadIdx.x;
  red[tid] = v; __syncthreads();
  for (int s = 128; s > 0; s >>= 1) {
    if (tid < s) red[tid] += red[tid + s];
    __syncthreads();
  }
  float r = red[0]; __syncthreads();
  return r;
}

__global__ void __launch_bounds__(256)
head_kernel(const float* __restrict__ out_f, const float* __restrict__ out_b,
            const int* __restrict__ entities, const int* __restrict__ slices,
            const float* __restrict__ slice_lens,
            const int* __restrict__ entity_masks, const int* __restrict__ slice_masks,
            const float* __restrict__ ent_W, const float* __restrict__ ent_bias,
            const float* __restrict__ att_w,
            const float* __restrict__ dense_W, const float* __restrict__ dense_b,
            float* __restrict__ out) {
  const int B = 64, H = 256, E = 8, S = 256, L = 19;
  const float BETA = 0.05f;
  int b = blockIdx.x, tid = threadIdx.x;

  __shared__ float eeL[E * H];
  __shared__ float pL[H];
  __shared__ float relaL[S];
  __shared__ float red[256];
  __shared__ float score[E + S];
  __shared__ float attL[H];
  __shared__ float ewL[E];

  // ee = enc[b, entities[b]] ; enc(b,t,h) = out_f[t,b,h] + out_b[t,b,h]
  for (int i = tid; i < E * H; i += 256) {
    int e = i / H, h = i % H;
    size_t base = ((size_t)entities[b * E + e] * B + b) * H;
    eeL[i] = out_f[base + h] + out_b[base + h];
  }
  __syncthreads();

  // entity gate softmax
  if (tid < E) {
    float s = ent_bias[0];
    for (int h = 0; h < H; ++h) s += eeL[tid * H + h] * ent_W[h];
    red[tid] = (entity_masks[b * E + tid] == 0) ? -INFINITY : s;
  }
  __syncthreads();
  if (tid == 0) {
    float mx = -INFINITY;
    for (int e = 0; e < E; ++e) mx = fmaxf(mx, red[e]);
    float sm = 0.f;
    for (int e = 0; e < E; ++e) { ewL[e] = expf(red[e] - mx); sm += ewL[e]; }
    for (int e = 0; e < E; ++e) ewL[e] /= sm;
  }
  __syncthreads();
  {
    float s = 0.f;
    for (int e = 0; e < E; ++e) s += ewL[e] * eeL[e * H + tid];
    pL[tid] = tanhf(s);
  }
  __syncthreads();

  // words softmax -> rela
  float w;
  {
    size_t base = ((size_t)slices[b * S + tid] * B + b) * H;
    float s = 0.f;
    for (int h = 0; h < H; ++h) s += (out_f[base + h] + out_b[base + h]) * pL[h];
    w = (slice_masks[b * S + tid] == 0) ? -INFINITY : s;
  }
  float mx = blk_max(w, red);
  float e  = expf(w - mx);                 // exp(-inf) == 0
  float sm = blk_sum(e, red);
  float ww = e / sm * slice_lens[b];
  float r1 = ((ww > BETA) ? ww : 0.f) / slice_lens[b];
  float mr = blk_max(r1, red);
  relaL[tid] = r1 / mr;
  __syncthreads();

  // attention scores over Hcat = [ee; rela*es]
  for (int n = tid; n < E + S; n += 256) {
    float acc = 0.f;
    if (n < E) {
      for (int h = 0; h < H; ++h) acc += tanhf(eeL[n * H + h]) * att_w[h];
    } else {
      int s2 = n - E;
      size_t base = ((size_t)slices[b * S + s2] * B + b) * H;
      float rl = relaL[s2];
      for (int h = 0; h < H; ++h)
        acc += tanhf(rl * (out_f[base + h] + out_b[base + h])) * att_w[h];
    }
    score[n] = (acc == 0.f) ? -INFINITY : acc;
  }
  __syncthreads();
  {
    float v0 = score[tid];
    float v1 = (tid < E) ? score[S + tid] : -INFINITY;
    float m2 = blk_max(fmaxf(v0, v1), red);
    float e0 = expf(v0 - m2);
    float e1 = (tid < E) ? expf(v1 - m2) : 0.f;
    float s2 = blk_sum(e0 + e1, red);
    score[tid] = e0 / s2;
    if (tid < E) score[S + tid] = e1 / s2;
  }
  __syncthreads();
  {
    float s = 0.f;
    for (int n = 0; n < E; ++n) s += score[n] * eeL[n * H + tid];
    for (int s2 = 0; s2 < S; ++s2) {
      size_t base = ((size_t)slices[b * S + s2] * B + b) * H;
      s += score[E + s2] * relaL[s2] * (out_f[base + tid] + out_b[base + tid]);
    }
    attL[tid] = s;
  }
  __syncthreads();
  if (tid < L) {
    float o = dense_b[tid];
    for (int h = 0; h < H; ++h) o += tanhf(attL[h]) * dense_W[tid * H + h];
    out[b * L + tid] = o;
  }
}

// ------------------------------ launcher -----------------------------------
extern "C" void kernel_launch(void* const* d_in, const int* in_sizes, int n_in,
                              void* d_out, int out_size, void* d_ws, size_t ws_size,
                              hipStream_t stream) {
  (void)in_sizes; (void)n_in; (void)out_size; (void)ws_size;
  const int B = 64, T = 1024, D = 256, H = 256;

  const int*   sents        = (const int*)d_in[0];
  const int*   lens         = (const int*)d_in[1];
  const int*   entities     = (const int*)d_in[2];
  const int*   slices       = (const int*)d_in[3];
  const float* slice_lens   = (const float*)d_in[4];
  const int*   entity_masks = (const int*)d_in[5];
  const int*   slice_masks  = (const int*)d_in[6];
  const float* emb          = (const float*)d_in[7];
  const float* Q            = (const float*)d_in[8];
  const float* R            = (const float*)d_in[9];
  const float* Wih_f        = (const float*)d_in[10];
  const float* Whh_f        = (const float*)d_in[11];
  const float* bih_f        = (const float*)d_in[12];
  const float* bhh_f        = (const float*)d_in[13];
  const float* Wih_b        = (const float*)d_in[14];
  const float* Whh_b        = (const float*)d_in[15];
  const float* bih_b        = (const float*)d_in[16];
  const float* bhh_b        = (const float*)d_in[17];
  const float* ent_W        = (const float*)d_in[18];
  const float* ent_b        = (const float*)d_in[19];
  const float* att_w        = (const float*)d_in[20];
  const float* dense_W      = (const float*)d_in[21];
  const float* dense_b      = (const float*)d_in[22];

  // workspace layout
  _Float16* xT   = (_Float16*)d_ws;
  _Float16* Qsw  = xT  + (size_t)T * B * D;            // 2 * 256*256 halves
  _Float16* Rsw  = Qsw + (size_t)2 * H * D;
  _Float16* WihS = Rsw + (size_t)2 * D * H;            // 2 * 256*768 halves
  _Float16* WhhS = WihS + (size_t)2 * D * 3 * H;
  float* out_f = (float*)(WhhS + (size_t)2 * H * 3 * H);
  float* out_b = out_f + (size_t)T * B * H;

  // Phase 0: embedding gather -> f16 xT [T,B,D]
  {
    int total = T * B * D;
    embed_kernel<<<(total + 255) / 256, 256, 0, stream>>>(sents, emb, xT);
  }

  // Phase 1: weight f16 conversion + WMMA B-fragment swizzle
  for (int dir = 0; dir < 2; ++dir) {
    swizzle_b_kernel<<<(H * D + 255) / 256, 256, 0, stream>>>(
        Q + (size_t)dir * H * D, Qsw + (size_t)dir * H * D, H, D, 0);
    swizzle_b_kernel<<<(D * H + 255) / 256, 256, 0, stream>>>(
        R + (size_t)dir * D * H, Rsw + (size_t)dir * D * H, D, H, 0);
    swizzle_b_kernel<<<(D * 3 * H + 255) / 256, 256, 0, stream>>>(
        dir ? Wih_b : Wih_f, WihS + (size_t)dir * D * 3 * H, D, 3 * H, 1);
    swizzle_b_kernel<<<(H * 3 * H + 255) / 256, 256, 0, stream>>>(
        dir ? Whh_b : Whh_f, WhhS + (size_t)dir * H * 3 * H, H, 3 * H, 1);
  }

  // Phase 2: persistent recurrent scan, one workgroup per direction
  size_t shmem = (size_t)3 * 16384 * sizeof(_Float16)  // x16, h16, hc16
               + (size_t)1536 * sizeof(float)          // biases
               + (size_t)64 * sizeof(int);             // mask
  hipFuncSetAttribute((const void*)scan_kernel,
                      hipFuncAttributeMaxDynamicSharedMemorySize, (int)shmem);
  scan_kernel<<<2, 512, shmem, stream>>>(xT, Qsw, Rsw, WihS, WhhS, lens,
                                         bih_f, bhh_f, bih_b, bhh_b,
                                         out_f, out_b);

  // Phase 3: gather + gated attention + dense head
  head_kernel<<<B, 256, 0, stream>>>(out_f, out_b, entities, slices, slice_lens,
                                     entity_masks, slice_masks, ent_W, ent_b,
                                     att_w, dense_W, dense_b, (float*)d_out);
}
